// SparseTransformer_14628658610293
// MI455X (gfx1250) — compile-verified, matrix operand order checked by reading, not run on hardware
//
#include <hip/hip_runtime.h>
#include <hip/hip_bf16.h>

// Problem constants (must match reference)
#define N_DIM  1024
#define M_DIM  4096
#define A_DIM  256
#define BATCH  8192
#define LN_EPS 1e-5f

typedef __bf16 bf16x16 __attribute__((ext_vector_type(16)));
typedef __bf16 bf16x8  __attribute__((ext_vector_type(8)));
typedef __bf16 bf16x4  __attribute__((ext_vector_type(4)));
typedef float  f32x8   __attribute__((ext_vector_type(8)));

// ---------------------------------------------------------------------------
// WMMA fragment loader (CDNA5 wave32, 16-bit A/B layout, CDNA5 ISA §7.12.2)
// Both operands of C = A * B^T are row-major [16 x K] with K contiguous, so A
// and B fragments use identical addressing:
//   lanes 0-15 : row = lane,      kb = 0,  K = {0..7} U {16..23}   (+k0)
//   lanes 16-31: row = lane - 16, kb = 8,  K = {8..15} U {24..31}  (+k0)
// ---------------------------------------------------------------------------
__device__ __forceinline__ bf16x16
load_frag_rowmajor(const __bf16* __restrict__ base, int ld, int row16, int k0, int lane)
{
    const int r  = lane & 15;
    const int kb = (lane >> 4) << 3;                 // 0 or 8
    const __bf16* p = base + (size_t)(row16 + r) * ld + k0 + kb;
    bf16x8 lo = *(const bf16x8*)(p);                 // K = kb .. kb+7
    bf16x8 hi = *(const bf16x8*)(p + 16);            // K = kb+16 .. kb+23
    return __builtin_shufflevector(lo, hi, 0,1,2,3,4,5,6,7,8,9,10,11,12,13,14,15);
}

__device__ __forceinline__ f32x8
wmma_bf16(bf16x16 a, bf16x16 b, f32x8 c)
{
    return __builtin_amdgcn_wmma_f32_16x16x32_bf16(false, a, false, b,
                                                   (short)0, c, false, false);
}

// ---------------------------------------------------------------------------
// Pre-pass: f32 -> bf16, vectorized (float4 in, 4x bf16 out)
// ---------------------------------------------------------------------------
__global__ void k_cvt_bf16(const float* __restrict__ in, __bf16* __restrict__ out, int n4)
{
    int i = blockIdx.x * blockDim.x + threadIdx.x;
    int stride = gridDim.x * blockDim.x;
    for (; i < n4; i += stride) {
        const float4 v = ((const float4*)in)[i];
        bf16x4 o;
        o[0] = (__bf16)v.x; o[1] = (__bf16)v.y;
        o[2] = (__bf16)v.z; o[3] = (__bf16)v.w;
        ((bf16x4*)out)[i] = o;
    }
}

// ---------------------------------------------------------------------------
// Kernel 1: q = LayerNorm(x @ Wq^T + b)  -> bf16   [BATCH, A_DIM]
// Block = 128 threads (4 waves), 16 batch rows per block.
//  - x tile (16 rows x 1024 = contiguous 32 KB) staged into LDS with
//    global_load_async_to_lds_b128 (ASYNCcnt) -> A fragments come from DS,
//    eliminating the 4x redundant per-wave A re-reads from global.
//  - Wq B-fragments double-buffered via manual unroll-by-2: no register
//    rotation copies, no WMMA->VALU hazard NOPs, loads stay a chunk ahead.
// ---------------------------------------------------------------------------
__global__ __launch_bounds__(128) void
k_query_ln(const __bf16* __restrict__ xb,    // [BATCH, N_DIM] bf16
           const __bf16* __restrict__ wqb,   // [A_DIM, N_DIM] bf16
           const float*  __restrict__ bias,  // [A_DIM]
           const float*  __restrict__ g,     // [A_DIM]
           const float*  __restrict__ be,    // [A_DIM]
           __bf16*       __restrict__ qb)    // [BATCH, A_DIM] bf16
{
    __shared__ __align__(16) __bf16 xs[16][N_DIM];   // 32 KB x tile
    __shared__ float hq[16][A_DIM];                  // 16 KB pre-LN activations
    __shared__ float mu_s[16], rs_s[16];

    const int tid  = threadIdx.x;
    const int wave = tid >> 5;
    const int lane = tid & 31;
    const int row0 = blockIdx.x * 16;

    // ---- async-stage the contiguous 32 KB x tile into LDS (ASYNCcnt path) ----
    {
        const char* gsrc = (const char*)(xb + (size_t)row0 * N_DIM);
        const unsigned lbase = (unsigned)(size_t)(&xs[0][0]);   // LDS byte offset
        #pragma unroll 4
        for (int c = tid; c < (16 * N_DIM * 2) / 16; c += 128) {
            const unsigned loff = lbase + c * 16;
            const char*    ga   = gsrc + c * 16;
            asm volatile("global_load_async_to_lds_b128 %0, %1, off"
                         :: "v"(loff), "v"(ga) : "memory");
        }
        asm volatile("s_wait_asynccnt 0" ::: "memory");
    }
    __syncthreads();

    const int rA  = lane & 15;
    const int kbA = (lane >> 4) << 3;
    const int nb  = wave * 4;                 // first of this wave's 4 n-tiles

    f32x8 acc[4] = {};
    bf16x16 b0[4], b1[4];
    #pragma unroll
    for (int t = 0; t < 4; ++t)
        b0[t] = load_frag_rowmajor(wqb, N_DIM, (nb + t) * 16, 0, lane);

    // Unroll-by-2 software pipeline (N_DIM is a multiple of 64).  The final
    // b0 prefetch at k0+64 == N_DIM reads 64 B past Wq_bf16 into workspace
    // padding (dead data, never used) — avoids a tail branch in the hot loop.
    for (int k0 = 0; k0 < N_DIM; k0 += 64) {
        #pragma unroll
        for (int t = 0; t < 4; ++t)
            b1[t] = load_frag_rowmajor(wqb, N_DIM, (nb + t) * 16, k0 + 32, lane);
        {
            bf16x8 lo = *(const bf16x8*)&xs[rA][k0 + kbA];
            bf16x8 hi = *(const bf16x8*)&xs[rA][k0 + kbA + 16];
            bf16x16 a = __builtin_shufflevector(lo, hi,
                            0,1,2,3,4,5,6,7,8,9,10,11,12,13,14,15);
            #pragma unroll
            for (int t = 0; t < 4; ++t)
                acc[t] = wmma_bf16(a, b0[t], acc[t]);
        }
        #pragma unroll
        for (int t = 0; t < 4; ++t)
            b0[t] = load_frag_rowmajor(wqb, N_DIM, (nb + t) * 16, k0 + 64, lane);
        {
            bf16x8 lo = *(const bf16x8*)&xs[rA][k0 + 32 + kbA];
            bf16x8 hi = *(const bf16x8*)&xs[rA][k0 + 32 + kbA + 16];
            bf16x16 a = __builtin_shufflevector(lo, hi,
                            0,1,2,3,4,5,6,7,8,9,10,11,12,13,14,15);
            #pragma unroll
            for (int t = 0; t < 4; ++t)
                acc[t] = wmma_bf16(a, b1[t], acc[t]);
        }
    }

    // Scatter C tiles (+bias) into LDS.  C layout: VGPR r, lane L -> M = r + 8*(L>=16), N = L&15.
    const int mrow = (lane >> 4) << 3;
    #pragma unroll
    for (int t = 0; t < 4; ++t) {
        const int col = (nb + t) * 16 + (lane & 15);
        const float bv = bias[col];
        #pragma unroll
        for (int r = 0; r < 8; ++r)
            hq[mrow + r][col] = acc[t][r] + bv;
    }
    __syncthreads();

    // Per-row LayerNorm statistics (16 rows x 256 cols — trivial cost)
    if (tid < 16) {
        float s = 0.f, s2 = 0.f;
        for (int c = 0; c < A_DIM; ++c) { float v = hq[tid][c]; s += v; s2 += v * v; }
        const float mu  = s * (1.0f / A_DIM);
        const float var = s2 * (1.0f / A_DIM) - mu * mu;
        mu_s[tid] = mu;
        rs_s[tid] = rsqrtf(var + LN_EPS);
    }
    __syncthreads();

    // Normalize, affine, convert to bf16, coalesced store.
    for (int e = tid; e < 16 * A_DIM; e += 128) {
        const int r = e >> 8, c = e & (A_DIM - 1);
        const float v = (hq[r][c] - mu_s[r]) * rs_s[r] * g[c] + be[c];
        qb[(size_t)(row0 + r) * A_DIM + c] = (__bf16)v;
    }
}

// ---------------------------------------------------------------------------
// Kernel 2 (fused): scores -> sparsemax -> sparse attn @ V for a 16-row block.
// Block = 256 threads (8 waves).  Scores tile 16 x 4096 f32 = 256 KB lives
// entirely in the 320 KB WGP LDS (CDNA5-only capacity).
// Each wave computes TWO 16x16 m-tiles per iteration with the K-chunk loop
// fully unrolled: 16 independent b128 load pairs + 16 WMMAs per iteration so
// VMEM latency overlaps the matrix pipe.  Sparsemax row scans use ds_load_b128.
// ---------------------------------------------------------------------------
__global__ __launch_bounds__(256) void
k_attn_sparse(const __bf16* __restrict__ qb,  // [BATCH, A_DIM] bf16
              const __bf16* __restrict__ kb,  // [M_DIM, A_DIM] bf16
              const float*  __restrict__ V,   // [M_DIM, N_DIM] f32
              float*        __restrict__ out) // [BATCH, N_DIM] f32
{
    __shared__ float sc[16][M_DIM];           // 256 KB score tile

    const int tid  = threadIdx.x;
    const int wave = tid >> 5;                // 0..7
    const int lane = tid & 31;
    const int row0 = blockIdx.x * 16;
    const float scale = 0.0625f;              // 1/sqrt(A_DIM)

    // Hoist all 8 A-fragments (full K = 256) of the q tile into registers.
    bf16x16 afrag[8];
    #pragma unroll
    for (int c = 0; c < 8; ++c)
        afrag[c] = load_frag_rowmajor(qb, A_DIM, row0, c * 32, lane);

    const int mrow = (lane >> 4) << 3;
    const int ncol = lane & 15;

    // scores = q @ K^T * scale; wave sweeps tile pairs (mt, mt+8).
    for (int mt = wave; mt < M_DIM / 16; mt += 16) {
        const int mt1 = mt + 8;
        f32x8 acc0 = {}, acc1 = {};
        #pragma unroll
        for (int c = 0; c < 8; ++c) {
            bf16x16 bA = load_frag_rowmajor(kb, A_DIM, mt  * 16, c * 32, lane);
            bf16x16 bB = load_frag_rowmajor(kb, A_DIM, mt1 * 16, c * 32, lane);
            acc0 = wmma_bf16(afrag[c], bA, acc0);
            acc1 = wmma_bf16(afrag[c], bB, acc1);
        }
        #pragma unroll
        for (int r = 0; r < 8; ++r) {
            float s0 = acc0[r] * scale;
            float s1 = acc1[r] * scale;
            s0 = (s0 < -1e10f) ? 0.0f : s0;   // reference's -1e10 masking
            s1 = (s1 < -1e10f) ? 0.0f : s1;
            sc[mrow + r][mt  * 16 + ncol] = s0;
            sc[mrow + r][mt1 * 16 + ncol] = s1;
        }
    }
    __syncthreads();

    // Each wave handles 2 rows: bisection sparsemax + sparse gather of V rows.
    #pragma unroll 1
    for (int rr = 0; rr < 2; ++rr) {
        const int r = wave * 2 + rr;
        const float*  __restrict__ srow  = &sc[r][0];
        const float4* __restrict__ srow4 = (const float4*)srow;   // 1024 vec4

        // Row max via ds_load_b128 (tau is in [max-1, max]: the support
        // contains the max element and sum over support of (s - tau) == 1).
        float mx = -3.4e38f;
        #pragma unroll 4
        for (int j = lane; j < M_DIM / 4; j += 32) {
            const float4 v = srow4[j];
            mx = fmaxf(mx, fmaxf(fmaxf(v.x, v.y), fmaxf(v.z, v.w)));
        }
        #pragma unroll
        for (int o = 16; o > 0; o >>= 1) mx = fmaxf(mx, __shfl_xor(mx, o, 32));

        // Bisection on g(tau) = sum(max(s - tau, 0)) - 1  (monotone decreasing).
        float lo = mx - 1.0f, hi = mx;
        for (int it = 0; it < 26; ++it) {
            const float tau = 0.5f * (lo + hi);
            float s = 0.f;
            #pragma unroll 4
            for (int j = lane; j < M_DIM / 4; j += 32) {
                const float4 v = srow4[j];
                s += fmaxf(v.x - tau, 0.f) + fmaxf(v.y - tau, 0.f)
                   + fmaxf(v.z - tau, 0.f) + fmaxf(v.w - tau, 0.f);
            }
            #pragma unroll
            for (int o = 16; o > 0; o >>= 1) s += __shfl_xor(s, o, 32);
            if (s > 1.0f) lo = tau; else hi = tau;
        }
        const float tau = 0.5f * (lo + hi);

        // Sparse accumulate: out_row = sum over nonzero p of p * V[m, :].
        // lane owns n = j*128 + lane*4 .. +3  (8 float4 accumulators).
        float4 accv[8];
        #pragma unroll
        for (int j = 0; j < 8; ++j) accv[j] = make_float4(0.f, 0.f, 0.f, 0.f);

        for (int m0 = 0; m0 < M_DIM; m0 += 32) {
            const float p = fmaxf(srow[m0 + lane] - tau, 0.f);
            // Prefetch the V rows this group will touch (global_prefetch_b8).
            if (p > 0.f)
                __builtin_prefetch(V + (size_t)(m0 + lane) * N_DIM, 0, 1);
            unsigned msk = (unsigned)__ballot(p > 0.f);   // wave32: low 32 bits
            while (msk) {
                const int i = __ffs(msk) - 1;
                msk &= msk - 1;
                const float pv = __shfl(p, i, 32);
                const float4* __restrict__ vrow =
                    (const float4*)(V + (size_t)(m0 + i) * N_DIM) + lane;
                #pragma unroll
                for (int j = 0; j < 8; ++j) {
                    const float4 v = vrow[j * 32];
                    accv[j].x += pv * v.x;
                    accv[j].y += pv * v.y;
                    accv[j].z += pv * v.z;
                    accv[j].w += pv * v.w;
                }
            }
        }

        float4* __restrict__ orow = (float4*)(out + (size_t)(row0 + r) * N_DIM) + lane;
        #pragma unroll
        for (int j = 0; j < 8; ++j) orow[j * 32] = accv[j];
    }
}

// ---------------------------------------------------------------------------
// Host launcher.  Workspace layout (bf16 staging, ~22.5 MB total):
//   [ 0,16MB)  x_bf16   [BATCH,N_DIM]
//   [16,16.5)  Wq_bf16  [A_DIM,N_DIM]   (+padding slack used by dead prefetch)
//   [17,19)    K_bf16   [M_DIM,A_DIM]
//   [20,24)    q_bf16   [BATCH,A_DIM]
// ---------------------------------------------------------------------------
extern "C" void kernel_launch(void* const* d_in, const int* in_sizes, int n_in,
                              void* d_out, int out_size, void* d_ws, size_t ws_size,
                              hipStream_t stream) {
    const float* x    = (const float*)d_in[0];
    const float* Wq_w = (const float*)d_in[1];
    const float* Wq_b = (const float*)d_in[2];
    const float* ln_g = (const float*)d_in[3];
    const float* ln_b = (const float*)d_in[4];
    const float* K    = (const float*)d_in[5];
    const float* V    = (const float*)d_in[6];
    float* out = (float*)d_out;

    char* ws = (char*)d_ws;
    __bf16* xb  = (__bf16*)(ws);
    __bf16* wqb = (__bf16*)(ws + (size_t)16 * 1024 * 1024);
    __bf16* kbb = (__bf16*)(ws + (size_t)17 * 1024 * 1024);
    __bf16* qb  = (__bf16*)(ws + (size_t)20 * 1024 * 1024);

    // bf16 staging pre-passes (vectorized: float4 per thread per step)
    k_cvt_bf16<<<2048, 256, 0, stream>>>(x,    xb,  (BATCH * N_DIM) / 4);
    k_cvt_bf16<<<256,  256, 0, stream>>>(Wq_w, wqb, (A_DIM * N_DIM) / 4);
    k_cvt_bf16<<<512,  256, 0, stream>>>(K,    kbb, (M_DIM * A_DIM) / 4);

    // q = LN(x @ Wq^T + b)
    k_query_ln<<<BATCH / 16, 128, 0, stream>>>(xb, wqb, Wq_b, ln_g, ln_b, qb);

    // fused scores -> sparsemax -> sparse attn @ V
    k_attn_sparse<<<BATCH / 16, 256, 0, stream>>>(qb, kbb, V, out);
}
